// HierarchicalGNN_14113262535305
// MI455X (gfx1250) — matrix operand order repaired and checked by reading, not compiled, not run on hardware
//
#include <hip/hip_runtime.h>
#include <hip/hip_bf16.h>
#include <math.h>

#define NP     200000
#define NPOS   320
#define NTEAM  32
#define FEAT   128
#define EMB    64
#define HID    128
#define EP2P   400000
#define EP2T   640
#define ECHEM  1000000

typedef __attribute__((ext_vector_type(16))) __bf16 v16bf;
typedef __attribute__((ext_vector_type(8)))  __bf16 v8bf;
typedef __attribute__((ext_vector_type(8)))  float  v8f;

// ---------------------------------------------------------------------------
// WMMA fragment loaders from LDS (bf16, row-major with leading dim ldk).
// A-matrix 16x32 bf16 (ISA 7.12.2): lane L holds M=L&15; elems 0..7 are
// k = base..base+7, elems 8..15 are k = base+16..base+23, base = (L&16)?8:0.
// B-matrix 32x16 bf16: lane L holds N=L&15; same k order as A (elems 0..7 at
// base, 8..15 at base+16, base = (L&16)?8:0). We store W transposed (N-major) so both loaders
// read contiguous 16B/32B runs -> ds_load_b128.
// ---------------------------------------------------------------------------
__device__ inline v16bf frag_a(const __bf16* As, int ldk, int m0, int k0, int lane) {
    int r  = m0 + (lane & 15);
    int kb = k0 + ((lane & 16) ? 8 : 0);
    const __bf16* p = As + r * ldk + kb;
    union { v16bf v; v8bf h[2]; } u;
    u.h[0] = *(const v8bf*)(p);
    u.h[1] = *(const v8bf*)(p + 16);
    return u.v;
}

__device__ inline v16bf frag_b(const __bf16* WT, int ldk, int n0, int k0, int lane) {
    int n  = n0 + (lane & 15);
    int kb = k0 + ((lane & 16) ? 8 : 0);
    const __bf16* p = WT + n * ldk + kb;
    union { v16bf v; v8bf h[2]; } u;
    u.h[0] = *(const v8bf*)(p);
    u.h[1] = *(const v8bf*)(p + 16);
    return u.v;
}

// ---------------------------------------------------------------------------
// Generic GEMM: C[M,N] = act(A[M,K] @ W[K,N] + bias), A/W f32 in global,
// converted to bf16 in LDS; WMMA bf16 -> f32. 128 threads, 64 rows/block.
// dyn LDS = (64*K + N*K)*2 bytes.
// ---------------------------------------------------------------------------
__global__ void gemm_bias_act(const float* __restrict__ A, const float* __restrict__ W,
                              const float* __restrict__ bias, float* __restrict__ C,
                              int M, int K, int N, int act) {
    extern __shared__ char smem[];
    __bf16* As = (__bf16*)smem;       // 64 x K
    __bf16* WT = As + 64 * K;         // N x K (transposed)
    const int tid = threadIdx.x, lane = tid & 31, wave = tid >> 5;
    const int rowBase = blockIdx.x * 64;

    for (int idx = tid; idx < 64 * K; idx += blockDim.x) {
        int r = idx / K, k = idx - r * K;
        int gr = rowBase + r;
        As[idx] = (gr < M) ? (__bf16)A[(long)gr * K + k] : (__bf16)0.0f;
    }
    for (int idx = tid; idx < N * K; idx += blockDim.x) {
        int n = idx / K, k = idx - n * K;
        WT[idx] = (__bf16)W[(long)k * N + n];
    }
    __syncthreads();

    const int m0 = wave * 16;
    for (int n0 = 0; n0 < N; n0 += 16) {
        v8f c = {};
        for (int k0 = 0; k0 < K; k0 += 32) {
            v16bf a = frag_a(As, K, m0, k0, lane);
            v16bf b = frag_b(WT, K, n0, k0, lane);
            c = __builtin_amdgcn_wmma_f32_16x16x32_bf16(false, a, false, b,
                                                        (short)0, c, false, false);
        }
        int n  = n0 + (lane & 15);
        int mh = (lane & 16) ? 8 : 0;
        float bv = bias[n];
#pragma unroll
        for (int r = 0; r < 8; ++r) {
            int gm = rowBase + m0 + r + mh;
            if (gm < M) {
                float v = c[r] + bv;
                if (act) v = fmaxf(v, 0.0f);
                C[(long)gm * N + n] = v;
            }
        }
    }
}

// ---------------------------------------------------------------------------
// Fused player encoder: OUT = (relu(X@W1+b1))@W2 + b2, all tiles in LDS.
// dyn LDS = (64*128 + 128*128 + 64*128 + 64*128)*2 = 80 KB (WGP has 320 KB).
// ---------------------------------------------------------------------------
__global__ void encoder_kernel(const float* __restrict__ X,  const float* __restrict__ W1,
                               const float* __restrict__ b1, const float* __restrict__ W2,
                               const float* __restrict__ b2, float* __restrict__ OUT, int M) {
    extern __shared__ char smem[];
    __bf16* Xs  = (__bf16*)smem;         // 64 x 128
    __bf16* W1T = Xs  + 64 * 128;        // 128 x 128
    __bf16* Hs  = W1T + 128 * 128;       // 64 x 128
    __bf16* W2T = Hs  + 64 * 128;        // 64 x 128
    const int tid = threadIdx.x, lane = tid & 31, wave = tid >> 5;
    const int rowBase = blockIdx.x * 64;

    for (int idx = tid; idx < 64 * 128; idx += blockDim.x) {
        int r = idx >> 7, k = idx & 127;
        int gr = rowBase + r;
        Xs[idx] = (gr < M) ? (__bf16)X[(long)gr * 128 + k] : (__bf16)0.0f;
    }
    for (int idx = tid; idx < 128 * 128; idx += blockDim.x) {
        int n = idx >> 7, k = idx & 127;
        W1T[idx] = (__bf16)W1[k * 128 + n];
    }
    for (int idx = tid; idx < 64 * 128; idx += blockDim.x) {
        int n = idx >> 7, k = idx & 127;
        W2T[idx] = (__bf16)W2[k * 64 + n];
    }
    __syncthreads();

    const int m0 = wave * 16;
    // H = relu(X @ W1 + b1), kept in LDS as bf16
    for (int n0 = 0; n0 < 128; n0 += 16) {
        v8f c = {};
        for (int k0 = 0; k0 < 128; k0 += 32) {
            c = __builtin_amdgcn_wmma_f32_16x16x32_bf16(
                    false, frag_a(Xs, 128, m0, k0, lane),
                    false, frag_b(W1T, 128, n0, k0, lane),
                    (short)0, c, false, false);
        }
        int n  = n0 + (lane & 15);
        int mh = (lane & 16) ? 8 : 0;
        float bv = b1[n];
#pragma unroll
        for (int r = 0; r < 8; ++r)
            Hs[(m0 + r + mh) * 128 + n] = (__bf16)fmaxf(c[r] + bv, 0.0f);
    }
    __syncthreads();
    // OUT = H @ W2 + b2
    for (int n0 = 0; n0 < 64; n0 += 16) {
        v8f c = {};
        for (int k0 = 0; k0 < 128; k0 += 32) {
            c = __builtin_amdgcn_wmma_f32_16x16x32_bf16(
                    false, frag_a(Hs, 128, m0, k0, lane),
                    false, frag_b(W2T, 128, n0, k0, lane),
                    (short)0, c, false, false);
        }
        int n  = n0 + (lane & 15);
        int mh = (lane & 16) ? 8 : 0;
        float bv = b2[n];
#pragma unroll
        for (int r = 0; r < 8; ++r) {
            int gm = rowBase + m0 + r + mh;
            if (gm < M) OUT[(long)gm * 64 + n] = c[r] + bv;
        }
    }
}

// --------------------------- attention passes ------------------------------
__device__ inline void atomicMaxFloat(float* addr, float val) {
    int old = __float_as_int(*addr);
    while (__int_as_float(old) < val) {
        int assumed = old;
        old = atomicCAS((int*)addr, assumed, __float_as_int(val));
        if (old == assumed) break;
    }
}

__global__ void edge_score_kernel(const int* __restrict__ src, const int* __restrict__ dst,
                                  const float* __restrict__ Qt, const float* __restrict__ Kt,
                                  float* __restrict__ scores, float* __restrict__ red, int E) {
    __shared__ float smax[8];
    int e = blockIdx.x * blockDim.x + threadIdx.x;
    float s = -1e30f;
    if (e < E) {
        const float* q = Qt + (long)dst[e] * EMB;
        const float* k = Kt + (long)src[e] * EMB;
        float acc = 0.0f;
#pragma unroll
        for (int j = 0; j < EMB; ++j) acc += q[j] * k[j];
        s = acc * 0.125f;     // 1/sqrt(64)
        scores[e] = s;
    }
    for (int off = 16; off > 0; off >>= 1) s = fmaxf(s, __shfl_down(s, off, 32));
    int lane = threadIdx.x & 31, wv = threadIdx.x >> 5;
    if (lane == 0) smax[wv] = s;
    __syncthreads();
    if (threadIdx.x == 0) {
        float m = smax[0];
        for (int i = 1; i < (int)(blockDim.x >> 5); ++i) m = fmaxf(m, smax[i]);
        atomicMaxFloat(&red[0], m);
    }
}

__global__ void edge_exp_kernel(float* __restrict__ scores, const float* __restrict__ red,
                                float* __restrict__ redSum, int E) {
    __shared__ float ssum[8];
    int e = blockIdx.x * blockDim.x + threadIdx.x;
    float mx = red[0];
    float v = 0.0f;
    if (e < E) {
        v = __expf(scores[e] - mx);
        scores[e] = v;
    }
    for (int off = 16; off > 0; off >>= 1) v += __shfl_down(v, off, 32);
    int lane = threadIdx.x & 31, wv = threadIdx.x >> 5;
    if (lane == 0) ssum[wv] = v;
    __syncthreads();
    if (threadIdx.x == 0) {
        float t = 0.0f;
        for (int i = 0; i < (int)(blockDim.x >> 5); ++i) t += ssum[i];
        atomicAdd(redSum, t);
    }
}

// emb[dst] += (expv[e]/sum) * Vt[src]; one thread per (edge, 8-feature slab)
__global__ void edge_msg_kernel(const int* __restrict__ src, const int* __restrict__ dst,
                                const float* __restrict__ expv, const float* __restrict__ Vt,
                                const float* __restrict__ redSum, float* __restrict__ emb, int E) {
    long gid = (long)blockIdx.x * blockDim.x + threadIdx.x;
    int e = (int)(gid >> 3), seg = (int)(gid & 7);
    if (e >= E) return;
    float w = expv[e] / redSum[0];
    const float* v = Vt + (long)src[e] * EMB + seg * 8;
    float* o = emb + (long)dst[e] * EMB + seg * 8;
#pragma unroll
    for (int j = 0; j < 8; ++j) atomicAdd(&o[j], w * v[j]);
}

// ------------------------- message-passing helpers -------------------------
__global__ void gather_emb_kernel(const int* __restrict__ idx, const float* __restrict__ table,
                                  float* __restrict__ out, int rows) {
    int i = blockIdx.x * blockDim.x + threadIdx.x;
    if (i < rows * EMB) {
        int r = i >> 6, c = i & 63;
        out[i] = table[idx[r] * EMB + c];
    }
}

// LDS-resident 320x64 accumulator (81 KB of the WGP's 320 KB LDS).
__global__ void p2p_scatter_kernel(const int* __restrict__ src, const int* __restrict__ dst,
                                   const float* __restrict__ Mt, float* __restrict__ agg,
                                   float* __restrict__ cnt, int E) {
    extern __shared__ float sm[];
    float* sAgg = sm;                  // NPOS*EMB
    float* sCnt = sm + NPOS * EMB;     // NPOS
    for (int i = threadIdx.x; i < NPOS * EMB + NPOS; i += blockDim.x) sm[i] = 0.0f;
    __syncthreads();
    for (long e = (long)blockIdx.x * blockDim.x + threadIdx.x; e < E;
         e += (long)gridDim.x * blockDim.x) {
        int d = dst[e];
        const float* m = Mt + (long)src[e] * EMB;
        float* o = sAgg + d * EMB;
#pragma unroll
        for (int j = 0; j < EMB; ++j) atomicAdd(&o[j], m[j]);
        atomicAdd(&sCnt[d], 1.0f);
    }
    __syncthreads();
    for (int i = threadIdx.x; i < NPOS * EMB; i += blockDim.x) atomicAdd(&agg[i], sAgg[i]);
    for (int i = threadIdx.x; i < NPOS; i += blockDim.x) atomicAdd(&cnt[i], sCnt[i]);
}

__global__ void p2t_scatter_kernel(const int* __restrict__ src, const int* __restrict__ dst,
                                   const float* __restrict__ Mt, float* __restrict__ agg,
                                   float* __restrict__ cnt, int E) {
    int e = blockIdx.x * blockDim.x + threadIdx.x;
    if (e >= E) return;
    int d = dst[e];
    const float* m = Mt + (long)src[e] * EMB;
#pragma unroll
    for (int j = 0; j < EMB; ++j) atomicAdd(&agg[d * EMB + j], m[j]);
    atomicAdd(&cnt[d], 1.0f);
}

__global__ void mean_div_kernel(float* __restrict__ agg, const float* __restrict__ cnt, int rows) {
    int i = blockIdx.x * blockDim.x + threadIdx.x;
    if (i < rows * EMB) agg[i] /= fmaxf(cnt[i >> 6], 1.0f);
}

__global__ void concat_kernel(const float* __restrict__ a, const float* __restrict__ b,
                              float* __restrict__ out, int rows) {
    int i = blockIdx.x * blockDim.x + threadIdx.x;
    if (i < rows * 2 * EMB) {
        int r = i >> 7, c = i & 127;
        out[i] = (c < EMB) ? a[r * EMB + c] : b[r * EMB + (c - EMB)];
    }
}

__global__ void init_kernel(float* aggPos, float* cntPos, float* aggTeam, float* cntTeam,
                            float* red) {
    int i = blockIdx.x * blockDim.x + threadIdx.x;
    if (i < NPOS * EMB)  aggPos[i]  = 0.0f;
    if (i < NPOS)        cntPos[i]  = 0.0f;
    if (i < NTEAM * EMB) aggTeam[i] = 0.0f;
    if (i < NTEAM)       cntTeam[i] = 0.0f;
    if (i == 0) { red[0] = -1e30f; red[1] = 0.0f; }
}

// ---------------------------- predictor head -------------------------------
__global__ void predictor_kernel(const float* __restrict__ team, const int* __restrict__ homeIdx,
                                 const int* __restrict__ awayIdx,
                                 const float* __restrict__ w1, const float* __restrict__ bias1,
                                 const float* __restrict__ w2, const float* __restrict__ bias2,
                                 const float* __restrict__ w3, const float* __restrict__ bias3,
                                 float* __restrict__ out) {
    __shared__ float ge[128];
    __shared__ float z1[128];
    __shared__ float z2[64];
    int t = threadIdx.x;
    int h = homeIdx[0], a = awayIdx[0];
    ge[t] = (t < 64) ? team[h * EMB + t] : team[a * EMB + (t - 64)];
    __syncthreads();
    {
        float acc = bias1[t];
        for (int k = 0; k < 128; ++k) acc += ge[k] * w1[k * 128 + t];
        z1[t] = fmaxf(acc, 0.0f);
    }
    __syncthreads();
    if (t < 64) {
        float acc = bias2[t];
        for (int k = 0; k < 128; ++k) acc += z1[k] * w2[k * 64 + t];
        z2[t] = fmaxf(acc, 0.0f);
    }
    __syncthreads();
    if (t == 0) {
        float acc = bias3[0];
        for (int k = 0; k < 64; ++k) acc += z2[k] * w3[k];
        out[0] = 1.0f / (1.0f + __expf(-acc));
    }
}

// ---------------------------------------------------------------------------
extern "C" void kernel_launch(void* const* d_in, const int* in_sizes, int n_in,
                              void* d_out, int out_size, void* d_ws, size_t ws_size,
                              hipStream_t stream) {
    const float* X        = (const float*)d_in[0];
    const int*   pos_idx  = (const int*)d_in[1];
    const int*   team_idx = (const int*)d_in[2];
    const int*   p2p_src  = (const int*)d_in[3];
    const int*   p2p_dst  = p2p_src + EP2P;
    const int*   p2t_src  = (const int*)d_in[4];
    const int*   p2t_dst  = p2t_src + EP2T;
    const int*   chem_src = (const int*)d_in[5];
    const int*   chem_dst = chem_src + ECHEM;
    const int*   home     = (const int*)d_in[6];
    const int*   away     = (const int*)d_in[7];
    const float* enc_w1 = (const float*)d_in[8];
    const float* enc_b1 = (const float*)d_in[9];
    const float* enc_w2 = (const float*)d_in[10];
    const float* enc_b2 = (const float*)d_in[11];
    const float* pos_tab  = (const float*)d_in[12];
    const float* team_tab = (const float*)d_in[13];
    const float* q_w = (const float*)d_in[14]; const float* q_b = (const float*)d_in[15];
    const float* k_w = (const float*)d_in[16]; const float* k_b = (const float*)d_in[17];
    const float* v_w = (const float*)d_in[18]; const float* v_b = (const float*)d_in[19];
    const float* p2p_msg_w = (const float*)d_in[20]; const float* p2p_msg_b = (const float*)d_in[21];
    const float* p2p_upd_w = (const float*)d_in[22]; const float* p2p_upd_b = (const float*)d_in[23];
    const float* p2t_msg_w = (const float*)d_in[24]; const float* p2t_msg_b = (const float*)d_in[25];
    const float* p2t_upd_w = (const float*)d_in[26]; const float* p2t_upd_b = (const float*)d_in[27];
    const float* gp_w1 = (const float*)d_in[28]; const float* gp_b1 = (const float*)d_in[29];
    const float* gp_w2 = (const float*)d_in[30]; const float* gp_b2 = (const float*)d_in[31];
    const float* gp_w3 = (const float*)d_in[32]; const float* gp_b3 = (const float*)d_in[33];
    float* out = (float*)d_out;

    // workspace layout (f32 elements)
    float* ws = (float*)d_ws;
    size_t off = 0;
    float* emb    = ws + off; off += (size_t)NP * EMB;   // attention result in-place
    float* Qt     = ws + off; off += (size_t)NP * EMB;   // reused as player msg table
    float* Kt     = ws + off; off += (size_t)NP * EMB;
    float* Vt     = ws + off; off += (size_t)NP * EMB;
    float* scores = ws + off; off += (size_t)ECHEM;      // reused as exp values
    float* red    = ws + off; off += 2;                  // [max, sum]
    float* posA   = ws + off; off += (size_t)NPOS * EMB;
    float* posB   = ws + off; off += (size_t)NPOS * EMB;
    float* aggPos = ws + off; off += (size_t)NPOS * EMB;
    float* cntPos = ws + off; off += NPOS;
    float* catBuf = ws + off; off += (size_t)NPOS * 2 * EMB; // also reused for pos-msg table
    float* teamA  = ws + off; off += (size_t)NTEAM * EMB;
    float* teamB  = ws + off; off += (size_t)NTEAM * EMB;
    float* aggTm  = ws + off; off += (size_t)NTEAM * EMB;
    float* cntTm  = ws + off; off += NTEAM;

    // 0) zero accumulators / reductions
    init_kernel<<<(NPOS * EMB + 255) / 256, 256, 0, stream>>>(aggPos, cntPos, aggTm, cntTm, red);

    // 1) fused player encoder -> emb
    {
        int blocks = (NP + 63) / 64;
        size_t lds = (size_t)(64 * 128 + 128 * 128 + 64 * 128 + 64 * 128) * sizeof(__bf16);
        encoder_kernel<<<blocks, 128, lds, stream>>>(X, enc_w1, enc_b1, enc_w2, enc_b2, emb, NP);
    }

    // 2) per-player Q/K/V tables (WMMA GEMMs)
    {
        int blocks = (NP + 63) / 64;
        size_t lds = (size_t)(64 * EMB + EMB * EMB) * sizeof(__bf16);
        gemm_bias_act<<<blocks, 128, lds, stream>>>(emb, q_w, q_b, Qt, NP, EMB, EMB, 0);
        gemm_bias_act<<<blocks, 128, lds, stream>>>(emb, k_w, k_b, Kt, NP, EMB, EMB, 0);
        gemm_bias_act<<<blocks, 128, lds, stream>>>(emb, v_w, v_b, Vt, NP, EMB, EMB, 0);
    }

    // 3) global softmax over edges: scores+max, exp+sum, weighted scatter into emb
    edge_score_kernel<<<(ECHEM + 255) / 256, 256, 0, stream>>>(chem_src, chem_dst, Qt, Kt,
                                                               scores, red, ECHEM);
    edge_exp_kernel<<<(ECHEM + 255) / 256, 256, 0, stream>>>(scores, red, red + 1, ECHEM);
    {
        long total = (long)ECHEM * 8;
        edge_msg_kernel<<<(unsigned)((total + 255) / 256), 256, 0, stream>>>(
            chem_src, chem_dst, scores, Vt, red + 1, emb, ECHEM);
    }

    // 4) player->position: message table is round-invariant
    {
        int blocks = (NP + 63) / 64;
        size_t lds = (size_t)(64 * EMB + EMB * EMB) * sizeof(__bf16);
        gemm_bias_act<<<blocks, 128, lds, stream>>>(emb, p2p_msg_w, p2p_msg_b, Qt, NP, EMB, EMB, 1);
    }
    gather_emb_kernel<<<(NPOS * EMB + 255) / 256, 256, 0, stream>>>(pos_idx, pos_tab, posA, NPOS);
    {
        size_t lds = (size_t)(NPOS * EMB + NPOS) * sizeof(float);  // 81.25 KB in LDS
        p2p_scatter_kernel<<<256, 256, lds, stream>>>(p2p_src, p2p_dst, Qt, aggPos, cntPos, EP2P);
    }
    mean_div_kernel<<<(NPOS * EMB + 255) / 256, 256, 0, stream>>>(aggPos, cntPos, NPOS);
    {
        float* cur = posA; float* nxt = posB;
        size_t lds = (size_t)(64 * 2 * EMB + EMB * 2 * EMB) * sizeof(__bf16);
        for (int r = 0; r < 3; ++r) {
            concat_kernel<<<(NPOS * 2 * EMB + 255) / 256, 256, 0, stream>>>(cur, aggPos, catBuf, NPOS);
            gemm_bias_act<<<(NPOS + 63) / 64, 128, lds, stream>>>(catBuf, p2p_upd_w, p2p_upd_b,
                                                                  nxt, NPOS, 2 * EMB, EMB, 1);
            float* t = cur; cur = nxt; nxt = t;
        }
        // final positions are in `cur` (= posB after 3 rounds)
        // 5) position->team: message table from final positions (reuse catBuf)
        size_t lds2 = (size_t)(64 * EMB + EMB * EMB) * sizeof(__bf16);
        gemm_bias_act<<<(NPOS + 63) / 64, 128, lds2, stream>>>(cur, p2t_msg_w, p2t_msg_b,
                                                               catBuf, NPOS, EMB, EMB, 1);
    }
    gather_emb_kernel<<<(NTEAM * EMB + 255) / 256, 256, 0, stream>>>(team_idx, team_tab, teamA, NTEAM);
    p2t_scatter_kernel<<<(EP2T + 255) / 256, 256, 0, stream>>>(p2t_src, p2t_dst, catBuf,
                                                               aggTm, cntTm, EP2T);
    mean_div_kernel<<<(NTEAM * EMB + 255) / 256, 256, 0, stream>>>(aggTm, cntTm, NTEAM);
    {
        float* cur = teamA; float* nxt = teamB;
        size_t lds = (size_t)(64 * 2 * EMB + EMB * 2 * EMB) * sizeof(__bf16);
        for (int r = 0; r < 3; ++r) {
            concat_kernel<<<(NTEAM * 2 * EMB + 255) / 256, 256, 0, stream>>>(cur, aggTm, catBuf, NTEAM);
            gemm_bias_act<<<1, 128, lds, stream>>>(catBuf, p2t_upd_w, p2t_upd_b,
                                                   nxt, NTEAM, 2 * EMB, EMB, 1);
            float* t = cur; cur = nxt; nxt = t;
        }
        // 6) predictor head
        predictor_kernel<<<1, 128, 0, stream>>>(cur, home, away, gp_w1, gp_b1, gp_w2, gp_b2,
                                                gp_w3, gp_b3, out);
    }
}
